// MPRelativeSelfMultiheadAttn_60344290509435
// MI455X (gfx1250) — matrix operand order repaired and checked by reading, not compiled, not run on hardware
//
#include <hip/hip_runtime.h>

// ---------------------------------------------------------------------------
// MPRelativeSelfMultiheadAttn for MI455X (gfx1250, wave32, WMMA).
// Precision: bf16 inputs to WMMA, f32 accumulation (reference is f32).
// ---------------------------------------------------------------------------

constexpr int kT  = 1024;
constexpr int kB  = 2;
constexpr int kE  = 1024;
constexpr int kH  = 16;
constexpr int kF  = 8;
constexpr int kD  = kE / kH;      // 64
constexpr int kR  = 2 * kT - 1;   // 2047
constexpr int kBH = kB * kH;      // 32

typedef __attribute__((ext_vector_type(16))) __bf16 v16bf;
typedef __attribute__((ext_vector_type(8)))  float  v8f;

union Frag {
    uint4 q[2];
    v16bf v;
};

__device__ __forceinline__ unsigned short f2bf(float f) {
    unsigned int u = __float_as_uint(f);
    unsigned int r = u + 0x7FFFu + ((u >> 16) & 1u);   // round-to-nearest-even
    return (unsigned short)(r >> 16);
}
__device__ __forceinline__ float bf2f(unsigned short s) {
    return __uint_as_float(((unsigned int)s) << 16);
}

__device__ __forceinline__ v8f wmma_bf16(v16bf a, v16bf b, v8f c) {
    return __builtin_amdgcn_wmma_f32_16x16x32_bf16(
        false, a, false, b, (short)0, c, false, false);
}

// A-matrix 16x32 bf16 fragment (lane m = l&15, K chunks [Kb,Kb+8) and [16+Kb,..))
__device__ __forceinline__ Frag loadA(const unsigned short* base, int row0,
                                      int ld, int k0) {
    int l = threadIdx.x & 31, hi = l >> 4, m = l & 15;
    const unsigned short* p = base + (size_t)(row0 + m) * ld + k0 + hi * 8;
    Frag f;
    f.q[0] = *(const uint4*)(p);
    f.q[1] = *(const uint4*)(p + 16);
    return f;
}
__device__ __forceinline__ Frag loadA_clamped(const unsigned short* base, int row0,
                                              int ld, int k0, int maxRow) {
    int l = threadIdx.x & 31, hi = l >> 4, m = l & 15;
    int r = row0 + m; r = r > maxRow ? maxRow : r;
    const unsigned short* p = base + (size_t)r * ld + k0 + hi * 8;
    Frag f;
    f.q[0] = *(const uint4*)(p);
    f.q[1] = *(const uint4*)(p + 16);
    return f;
}
// B-matrix 32x16 bf16 fragment: lane n = l&15, K chunks [Kb,Kb+8) and [16+Kb,..), Kb = (l>>4)*8
__device__ __forceinline__ Frag loadB(const unsigned short* base, int col0,
                                      int ld, int k0) {
    int l = threadIdx.x & 31, hi = l >> 4, n = l & 15;
    const unsigned short* p = base + (size_t)(col0 + n) * ld + k0 + hi * 8;
    Frag f;
    f.q[0] = *(const uint4*)(p);
    f.q[1] = *(const uint4*)(p + 16);
    return f;
}
__device__ __forceinline__ Frag loadB_clamped(const unsigned short* base, int col0,
                                              int ld, int k0, int maxCol) {
    int l = threadIdx.x & 31, hi = l >> 4, n = l & 15;
    int c = col0 + n; c = c > maxCol ? maxCol : c; c = c < 0 ? 0 : c;
    const unsigned short* p = base + (size_t)c * ld + k0 + hi * 8;
    Frag f;
    f.q[0] = *(const uint4*)(p);
    f.q[1] = *(const uint4*)(p + 16);
    return f;
}

// ---------------------------------------------------------------------------
// Hypernetwork: out[i] = dot(W[i*8 .. i*8+7], factor[0..7])
// ---------------------------------------------------------------------------
__global__ __launch_bounds__(256) void hyper_bf16_kernel(
    const float* __restrict__ W, const float* __restrict__ factor,
    unsigned short* __restrict__ dst, int n) {
    int i = blockIdx.x * 256 + threadIdx.x;
    if (i >= n) return;
    const float* p = W + (size_t)i * 8;
    float4 a = *(const float4*)p;
    float4 b = *(const float4*)(p + 4);
    float s = a.x * factor[0] + a.y * factor[1] + a.z * factor[2] + a.w * factor[3]
            + b.x * factor[4] + b.y * factor[5] + b.z * factor[6] + b.w * factor[7];
    dst[i] = f2bf(s);
}
__global__ __launch_bounds__(256) void hyper_f32_kernel(
    const float* __restrict__ W, const float* __restrict__ factor,
    float* __restrict__ dst, int n) {
    int i = blockIdx.x * 256 + threadIdx.x;
    if (i >= n) return;
    const float* p = W + (size_t)i * 8;
    float4 a = *(const float4*)p;
    float4 b = *(const float4*)(p + 4);
    dst[i] = a.x * factor[0] + a.y * factor[1] + a.z * factor[2] + a.w * factor[3]
           + b.x * factor[4] + b.y * factor[5] + b.z * factor[6] + b.w * factor[7];
}

__global__ __launch_bounds__(256) void cvt_bf16_kernel(
    const float* __restrict__ src, unsigned short* __restrict__ dst, int n) {
    int i = blockIdx.x * 256 + threadIdx.x;
    if (i >= n) return;
    dst[i] = f2bf(src[i]);
}

// ---------------------------------------------------------------------------
// Generic WMMA GEMM: C[M,N] = A[M,K](bf16) * B[N,K](bf16)^T + bias[N]
// 256 threads = 8 waves. Block tile 64(M) x 256(N); each wave computes a
// 32x64 tile (2 A-frags x 4 B-frags -> 8 WMMAs per 32-wide k-step), which
// amortizes fragment traffic to ~11 MACs/byte from cache.
// ---------------------------------------------------------------------------
template <bool F32OUT>
__global__ __launch_bounds__(256) void gemm_kernel(
    const unsigned short* __restrict__ A, const unsigned short* __restrict__ Bm,
    const float* __restrict__ bias, void* __restrict__ Cout,
    int Mdim, int Ndim, int Kdim) {
    int l = threadIdx.x & 31, w = threadIdx.x >> 5;
    int mT = (Mdim + 63) >> 6;
    int bm = blockIdx.x % mT, bn = blockIdx.x / mT;
    int m0 = bm * 64 + (w & 1) * 32;     // wave: rows [m0, m0+32)
    int n0 = bn * 256 + (w >> 1) * 64;   // wave: cols [n0, n0+64)
    int hi = l >> 4, n = l & 15;

    v8f acc[2][4];
#pragma unroll
    for (int mi = 0; mi < 2; ++mi)
#pragma unroll
        for (int j = 0; j < 4; ++j)
            acc[mi][j] = (v8f){0.f, 0.f, 0.f, 0.f, 0.f, 0.f, 0.f, 0.f};

    for (int k0 = 0; k0 < Kdim; k0 += 32) {
        Frag a0 = loadA_clamped(A, m0,      Kdim, k0, Mdim - 1);
        Frag a1 = loadA_clamped(A, m0 + 16, Kdim, k0, Mdim - 1);
#pragma unroll
        for (int j = 0; j < 4; ++j) {
            Frag b = loadB(Bm, n0 + j * 16, Kdim, k0);
            acc[0][j] = wmma_bf16(a0.v, b.v, acc[0][j]);
            acc[1][j] = wmma_bf16(a1.v, b.v, acc[1][j]);
        }
    }
#pragma unroll
    for (int j = 0; j < 4; ++j) {
        float bv = bias[n0 + j * 16 + n];
#pragma unroll
        for (int mi = 0; mi < 2; ++mi) {
#pragma unroll
            for (int g = 0; g < 8; ++g) {
                int m = m0 + mi * 16 + g + hi * 8;
                if (m < Mdim) {
                    float v = acc[mi][j][g] + bv;
                    if (F32OUT)
                        ((float*)Cout)[(size_t)m * Ndim + n0 + j * 16 + n] = v;
                    else
                        ((unsigned short*)Cout)[(size_t)m * Ndim + n0 + j * 16 + n] = f2bf(v);
                }
            }
        }
    }
}

// ---------------------------------------------------------------------------
// Split qkv [T,B,3E] -> qrw/qrr/k [BH,T,D] and vT [BH,D,T]
// ---------------------------------------------------------------------------
__global__ __launch_bounds__(256) void split_qkv_kernel(
    const unsigned short* __restrict__ qkv, const float* __restrict__ rw,
    const float* __restrict__ rr, unsigned short* __restrict__ qrw,
    unsigned short* __restrict__ qrr, unsigned short* __restrict__ kk,
    unsigned short* __restrict__ vT) {
    int idx = blockIdx.x * 256 + threadIdx.x;        // [0, BH*T*D)
    int d  = idx & (kD - 1);
    int t  = (idx >> 6) & (kT - 1);
    int bh = idx >> 16;
    int b = bh >> 4, h = bh & 15;
    int e = h * kD + d;
    size_t base = (size_t)t * (kB * 3 * kE) + (size_t)b * (3 * kE);
    float qv = bf2f(qkv[base + e]);
    qrw[idx] = f2bf(qv + rw[e]);
    qrr[idx] = f2bf(qv + rr[e]);
    kk[idx]  = qkv[base + kE + e];
    vT[(size_t)bh * kD * kT + (size_t)d * kT + t] = qkv[base + 2 * kE + e];
}

// posproj [R,B,E] -> rmat [BH,R,D]
__global__ __launch_bounds__(256) void split_r_kernel(
    const unsigned short* __restrict__ posproj, unsigned short* __restrict__ rmat) {
    int idx = blockIdx.x * 256 + threadIdx.x;        // [0, BH*R*D)
    if (idx >= kBH * kR * kD) return;
    int d  = idx & (kD - 1);
    int c  = (idx >> 6) % kR;
    int bh = idx / (kR * kD);
    int b = bh >> 4, h = bh & 15;
    rmat[idx] = posproj[(size_t)c * (kB * kE) + (size_t)b * kE + h * kD + d];
}

// ---------------------------------------------------------------------------
// Fused flash attention with rel-shift folded into the BD WMMA via per-half
// lane rotation of the accumulator tiles (no [B,H,T,R] materialization).
// One wave owns a 16-row query tile; 4 waves / block.
// ---------------------------------------------------------------------------
__global__ __launch_bounds__(128) void flash_attn_kernel(
    const unsigned short* __restrict__ qrw, const unsigned short* __restrict__ qrr,
    const unsigned short* __restrict__ kk,  const unsigned short* __restrict__ vT,
    const unsigned short* __restrict__ rmat, unsigned short* __restrict__ ao) {
    __shared__ __align__(16) unsigned short pbuf[4][16 * 32];

    const int l  = threadIdx.x & 31;
    const int w  = threadIdx.x >> 5;
    const int wid = blockIdx.x * 4 + w;
    const int bh = wid >> 6;              // 64 query tiles per (b,h)
    const int i0 = (wid & 63) << 4;
    const int b = bh >> 4, h = bh & 15;
    const int hi = l >> 4, n = l & 15;

    const unsigned short* q1 = qrw  + (size_t)bh * kT * kD;
    const unsigned short* q2 = qrr  + (size_t)bh * kT * kD;
    const unsigned short* kb = kk   + (size_t)bh * kT * kD;
    const unsigned short* vb = vT   + (size_t)bh * kD * kT;
    const unsigned short* rb = rmat + (size_t)bh * kR * kD;

    // Query fragments (j-invariant): K=64 -> 2 WMMA k-steps each
    Frag aw0 = loadA(q1, i0, kD, 0),  aw1 = loadA(q1, i0, kD, 32);
    Frag ar0 = loadA(q2, i0, kD, 0),  ar1 = loadA(q2, i0, kD, 32);

    v8f o[4];
#pragma unroll
    for (int db = 0; db < 4; ++db) o[db] = (v8f){0.f,0.f,0.f,0.f,0.f,0.f,0.f,0.f};
    float mrow[8], lrow[8];
#pragma unroll
    for (int g = 0; g < 8; ++g) { mrow[g] = -1e30f; lrow[g] = 0.f; }

    const float SC = 0.125f * 1.44269504088896f;  // 1/sqrt(64) * log2(e)

    for (int j0 = 0; j0 < kT; j0 += 32) {
        float t[2][8];
#pragma unroll
        for (int sub = 0; sub < 2; ++sub) {
            const int js = j0 + sub * 16;
            // AC = (q+rw) . k
            Frag bk0 = loadB(kb, js, kD, 0), bk1 = loadB(kb, js, kD, 32);
            v8f ac = {0.f,0.f,0.f,0.f,0.f,0.f,0.f,0.f};
            ac = wmma_bf16(aw0.v, bk0.v, ac);
            ac = wmma_bf16(aw1.v, bk1.v, ac);
            // BD band: r rows [c0, c0+32) cover all cc = jj-ii+15 in [0,30]
            const int c0 = js + (kT - 1) - i0 - 15;   // >= 0 always
            Frag b00 = loadB_clamped(rb, c0,      kD, 0,  kR - 1);
            Frag b01 = loadB_clamped(rb, c0,      kD, 32, kR - 1);
            Frag b10 = loadB_clamped(rb, c0 + 16, kD, 0,  kR - 1);
            Frag b11 = loadB_clamped(rb, c0 + 16, kD, 32, kR - 1);
            v8f p0 = {0.f,0.f,0.f,0.f,0.f,0.f,0.f,0.f};
            v8f p1 = {0.f,0.f,0.f,0.f,0.f,0.f,0.f,0.f};
            p0 = wmma_bf16(ar0.v, b00.v, p0);
            p0 = wmma_bf16(ar1.v, b01.v, p0);
            p1 = wmma_bf16(ar0.v, b10.v, p1);
            p1 = wmma_bf16(ar1.v, b11.v, p1);
            // Diagonal extract: within a 16-lane half, row ii is constant, so
            // the needed column cc = jj - ii + 15 is a uniform lane rotation.
#pragma unroll
            for (int g = 0; g < 8; ++g) {
                int iit = g + hi * 8;
                int s   = n + 15 - iit;               // in [0,30]
                int idx = (l & 16) | (s & 15);
                float v0 = __shfl(p0[g], idx, 32);
                float v1 = __shfl(p1[g], idx, 32);
                float bd = (s < 16) ? v0 : v1;
                t[sub][g] = (ac[g] + bd) * SC;
            }
        }
        // Online softmax over the 32 new columns
        float alpha[8], p[2][8];
#pragma unroll
        for (int g = 0; g < 8; ++g) {
            float rm = fmaxf(t[0][g], t[1][g]);
#pragma unroll
            for (int off = 8; off; off >>= 1) rm = fmaxf(rm, __shfl_xor(rm, off, 32));
            float mn = fmaxf(mrow[g], rm);
            alpha[g] = exp2f(mrow[g] - mn);
            mrow[g] = mn;
        }
#pragma unroll
        for (int g = 0; g < 8; ++g) {
            p[0][g] = exp2f(t[0][g] - mrow[g]);
            p[1][g] = exp2f(t[1][g] - mrow[g]);
            float rs = p[0][g] + p[1][g];
#pragma unroll
            for (int off = 8; off; off >>= 1) rs += __shfl_xor(rs, off, 32);
            lrow[g] = lrow[g] * alpha[g] + rs;
        }
#pragma unroll
        for (int db = 0; db < 4; ++db)
#pragma unroll
            for (int g = 0; g < 8; ++g) o[db][g] *= alpha[g];

        // Stage probs through LDS to re-layout C-matrix -> A-matrix fragment
#pragma unroll
        for (int sub = 0; sub < 2; ++sub)
#pragma unroll
            for (int g = 0; g < 8; ++g) {
                int mrow_t = g + hi * 8;
                pbuf[w][mrow_t * 32 + sub * 16 + n] = f2bf(p[sub][g]);
            }
        asm volatile("s_wait_dscnt 0" ::: "memory");
        Frag pa;
        {
            const unsigned short* pp = &pbuf[w][(l & 15) * 32 + hi * 8];
            pa.q[0] = *(const uint4*)(pp);
            pa.q[1] = *(const uint4*)(pp + 16);
        }
        // PV: O[16 x 64] += P[16 x 32] * V[32 x 64]
#pragma unroll
        for (int db = 0; db < 4; ++db) {
            Frag bv = loadB(vb, db * 16, kT, j0);   // vT[d, j] : K contiguous
            o[db] = wmma_bf16(pa.v, bv.v, o[db]);
        }
    }
    // Finalize and scatter to ao [T,B,E] (bf16, row-major for out-proj GEMM)
#pragma unroll
    for (int g = 0; g < 8; ++g) {
        float inv = 1.0f / lrow[g];
        int i = i0 + g + hi * 8;
#pragma unroll
        for (int db = 0; db < 4; ++db) {
            int d = db * 16 + n;
            ao[(size_t)i * (kB * kE) + (size_t)b * kE + h * kD + d] =
                f2bf(o[db][g] * inv);
        }
    }
}

// ---------------------------------------------------------------------------
// Host-side launch
// ---------------------------------------------------------------------------
static constexpr size_t alUp(size_t x) { return (x + 255) & ~(size_t)255; }

extern "C" void kernel_launch(void* const* d_in, const int* in_sizes, int n_in,
                              void* d_out, int out_size, void* d_ws, size_t ws_size,
                              hipStream_t stream) {
    (void)in_sizes; (void)n_in; (void)out_size; (void)ws_size;
    const float* input   = (const float*)d_in[0];
    const float* pos     = (const float*)d_in[1];
    const float* factor  = (const float*)d_in[2];
    const float* Win_w   = (const float*)d_in[3];
    const float* Wpos_w  = (const float*)d_in[4];
    const float* Wout_w  = (const float*)d_in[5];
    const float* bin_w   = (const float*)d_in[6];
    const float* bpos_w  = (const float*)d_in[7];
    const float* bout_w  = (const float*)d_in[8];
    const float* rw_w    = (const float*)d_in[9];
    const float* rr_w    = (const float*)d_in[10];

    char* ws = (char*)d_ws;
    size_t off = 0;
    auto alloc = [&](size_t bytes) { size_t o = off; off += alUp(bytes); return o; };

    unsigned short* Win   = (unsigned short*)(ws + alloc((size_t)3 * kE * kE * 2));
    unsigned short* Wpos  = (unsigned short*)(ws + alloc((size_t)kE * kE * 2));
    unsigned short* Wout  = (unsigned short*)(ws + alloc((size_t)kE * kE * 2));
    float* b_in  = (float*)(ws + alloc((size_t)3 * kE * 4));
    float* b_pos = (float*)(ws + alloc((size_t)kE * 4));
    float* b_out = (float*)(ws + alloc((size_t)kE * 4));
    float* rwv   = (float*)(ws + alloc((size_t)kE * 4));
    float* rrv   = (float*)(ws + alloc((size_t)kE * 4));
    unsigned short* in_bf   = (unsigned short*)(ws + alloc((size_t)kT * kB * kE * 2));
    unsigned short* pos_bf  = (unsigned short*)(ws + alloc((size_t)kR * kB * kE * 2));
    unsigned short* qkv_bf  = (unsigned short*)(ws + alloc((size_t)kT * kB * 3 * kE * 2));
    unsigned short* pp_bf   = (unsigned short*)(ws + alloc((size_t)kR * kB * kE * 2));
    unsigned short* qrw     = (unsigned short*)(ws + alloc((size_t)kBH * kT * kD * 2));
    unsigned short* qrr     = (unsigned short*)(ws + alloc((size_t)kBH * kT * kD * 2));
    unsigned short* kkv     = (unsigned short*)(ws + alloc((size_t)kBH * kT * kD * 2));
    unsigned short* vTv     = (unsigned short*)(ws + alloc((size_t)kBH * kD * kT * 2));
    unsigned short* rmat    = (unsigned short*)(ws + alloc((size_t)kBH * kR * kD * 2));
    unsigned short* ao      = (unsigned short*)(ws + alloc((size_t)kT * kB * kE * 2));

    // 1. Hypernetwork weight generation
    hyper_bf16_kernel<<<(3 * kE * kE + 255) / 256, 256, 0, stream>>>(Win_w, factor, Win, 3 * kE * kE);
    hyper_bf16_kernel<<<(kE * kE + 255) / 256, 256, 0, stream>>>(Wpos_w, factor, Wpos, kE * kE);
    hyper_bf16_kernel<<<(kE * kE + 255) / 256, 256, 0, stream>>>(Wout_w, factor, Wout, kE * kE);
    hyper_f32_kernel<<<(3 * kE + 255) / 256, 256, 0, stream>>>(bin_w, factor, b_in, 3 * kE);
    hyper_f32_kernel<<<(kE + 255) / 256, 256, 0, stream>>>(bpos_w, factor, b_pos, kE);
    hyper_f32_kernel<<<(kE + 255) / 256, 256, 0, stream>>>(bout_w, factor, b_out, kE);
    hyper_f32_kernel<<<(kE + 255) / 256, 256, 0, stream>>>(rw_w, factor, rwv, kE);
    hyper_f32_kernel<<<(kE + 255) / 256, 256, 0, stream>>>(rr_w, factor, rrv, kE);

    // 2. bf16 conversion of activations
    cvt_bf16_kernel<<<(kT * kB * kE + 255) / 256, 256, 0, stream>>>(input, in_bf, kT * kB * kE);
    cvt_bf16_kernel<<<(kR * kB * kE + 255) / 256, 256, 0, stream>>>(pos, pos_bf, kR * kB * kE);

    // 3. Projections (WMMA GEMM, block tile 64x256, wave tile 32x64)
    {   // qkv: [2048, 3072] = in_bf[2048,1024] * Win[3072,1024]^T + b_in
        int M = kT * kB, N = 3 * kE, K = kE;
        int mT = (M + 63) >> 6, nT = N >> 8;
        gemm_kernel<false><<<mT * nT, 256, 0, stream>>>(in_bf, Win, b_in, qkv_bf, M, N, K);
    }
    {   // pos proj: [4094, 1024]
        int M = kR * kB, N = kE, K = kE;
        int mT = (M + 63) >> 6, nT = N >> 8;
        gemm_kernel<false><<<mT * nT, 256, 0, stream>>>(pos_bf, Wpos, b_pos, pp_bf, M, N, K);
    }

    // 4. Head-major re-layouts (+ rw/rr bias add)
    split_qkv_kernel<<<(kBH * kT * kD) / 256, 256, 0, stream>>>(qkv_bf, rwv, rrv, qrw, qrr, kkv, vTv);
    split_r_kernel<<<(kBH * kR * kD + 255) / 256, 256, 0, stream>>>(pp_bf, rmat);

    // 5. Fused flash attention with folded rel-shift
    flash_attn_kernel<<<(kBH * (kT / 16)) / 4, 128, 0, stream>>>(qrw, qrr, kkv, vTv, rmat, ao);

    // 6. Output projection (f32 straight into d_out)
    {
        int M = kT * kB, N = kE, K = kE;
        int mT = (M + 63) >> 6, nT = N >> 8;
        gemm_kernel<true><<<mT * nT, 256, 0, stream>>>(ao, Wout, b_out, (float*)d_out, M, N, K);
    }
}
